// TransformerEncoder_88038239633464
// MI455X (gfx1250) — matrix operand order repaired and checked by reading, not compiled, not run on hardware
//
#include <hip/hip_runtime.h>
#include <math.h>

// ---- Shapes (fixed by the reference) ----
#define LL   6
#define DD   1024
#define HH   16
#define DFF  4096
#define BB   2
#define SS   2048
#define DEPTH 64
#define MM   (BB*SS)        // 4096 token rows

typedef __attribute__((ext_vector_type(16))) __bf16        v16bf;
typedef __attribute__((ext_vector_type(8)))  float         v8f;
typedef __attribute__((ext_vector_type(4)))  unsigned int  u32x4;
typedef __attribute__((ext_vector_type(4)))  int           i32x4;
typedef __attribute__((ext_vector_type(8)))  int           i32x8;

#define LP 40   // LDS row stride in halves: 80B = multiple of 16B (aligned b128), spreads banks

#if __has_builtin(__builtin_amdgcn_tensor_load_to_lds) && __has_builtin(__builtin_amdgcn_s_wait_tensorcnt)
#define HAVE_TDM 1
#endif

// ============================================================================
// Generic bf16 WMMA GEMM:  C = relu(scale * (A @ B) + bias)
//   A : bf16, M x K, row-major (lda)
//   BMODE 0: B fp32  K x N row-major (ldb = row stride)   [weights]
//   BMODE 1: B bf16  N x K row-major (ldb = row stride)   [Q@K^T, B transposed]
//   BMODE 2: B bf16  K x N row-major (ldb = row stride)   [P@V, needs N guard]
//   Cf: optional fp32 out (ldcf); Cb: optional bf16 out (ldcb); QKV -> head-major scatter
// Tile: 128x128x32. 256 threads = 8 wave32 waves; wave (wm 0..3, wn 0..1) owns 32x64
// = 2x4 v_wmma_f32_16x16x32_bf16 accumulators.
// ============================================================================
template <int BMODE, bool NG, bool QKV>
__global__ __launch_bounds__(256) void gemm_wmma_bf16(
    const __bf16* __restrict__ A, int lda,
    const void* __restrict__ Bp, long long ldb,
    const float* __restrict__ bias,
    float* __restrict__ Cf, int ldcf,
    __bf16* __restrict__ Cb, int ldcb,
    int M, int N, int K, float scale, int relu)
{
  __shared__ __align__(16) __bf16 As[128 * LP];
  __shared__ __align__(16) __bf16 Bs[128 * LP];

  const int tid = threadIdx.x;
  const int rowBase = blockIdx.y * 128;
  const int colBase = blockIdx.x * 128;
  const int wave = tid >> 5, lane = tid & 31;
  const int wm = wave & 3, wn = wave >> 2;
  const int half = lane >> 4, l16 = lane & 15;

  const float*  Bf = (const float*)Bp;
  const __bf16* Bb = (const __bf16*)Bp;

  v8f acc[2][4] = {};

  // global->LDS assignments
  const int am = tid >> 1;            // A row (0..127), each thread 16 contiguous k
  const int ak = (tid & 1) * 16;
  const int bn = (tid * 16) & 127;    // BMODE 0/2: 16 contiguous n at fixed k
  const int bk = tid >> 3;
  const int tn1 = tid & 127;          // BMODE 1: one n row, 16 contiguous k
  const int tk1 = (tid >> 7) * 16;

  for (int k0 = 0; k0 < K; k0 += 32) {
    // ---- A tile 128x32 -> As[m*LP + k], two aligned 16B chunks per thread ----
    {
      const u32x4* pa = (const u32x4*)(A + (size_t)(rowBase + am) * (size_t)lda + k0 + ak);
      u32x4 c0 = pa[0], c1 = pa[1];
      u32x4* da = (u32x4*)&As[am * LP + ak];
      da[0] = c0; da[1] = c1;
      if (k0 + 32 < K)  // prefetch next K-tile of A (global_prefetch_b8)
        __builtin_prefetch((const void*)(A + (size_t)(rowBase + am) * (size_t)lda + k0 + 32 + ak), 0, 1);
    }
    // ---- B tile 32x128 -> Bs[n*LP + k] (N-major so frag reads are b128) ----
    if constexpr (BMODE == 0) {
      // fp32 weights: 16 contiguous floats (64B) per thread, cvt to bf16
      const float* src = Bf + (size_t)(k0 + bk) * (size_t)ldb + colBase + bn;
      #pragma unroll
      for (int e = 0; e < 16; e++)
        Bs[(bn + e) * LP + bk] = (__bf16)src[e];
    } else if constexpr (BMODE == 1) {
#ifdef HAVE_TDM
      // Tensor Data Mover: 2D tile (32 x 128) of bf16, LDS rows padded 64B->80B.
      if (tid == 0) {
        unsigned lds_off = (unsigned)(uintptr_t)(&Bs[0]);
        unsigned long long ga =
            (unsigned long long)(uintptr_t)(Bb + (size_t)colBase * (size_t)ldb + k0);
        u32x4 g0;
        g0[0] = 1u;                                            // count=1, user descriptor
        g0[1] = lds_off;                                       // lds_addr
        g0[2] = (unsigned)(ga & 0xFFFFFFFFu);                  // global_addr[31:0]
        g0[3] = (unsigned)((ga >> 32) & 0x01FFFFFFu) | (2u << 30); // addr[56:32] | type=2
        i32x8 g1;
        g1[0] = (1 << 16) | (1 << 20) | (3 << 22) | (3 << 25); // data_size=2B, pad_en,
                                                               // pad_interval=16dw, pad=4dw
        g1[1] = (int)(32u << 16);                              // tensor_dim0 = 32
        g1[2] = (int)(128u << 16);                             // tensor_dim1 = 128
        g1[3] = (int)(32u << 16);                              // tile_dim0 = 32
        g1[4] = 128;                                           // tile_dim1 = 128
        g1[5] = (int)ldb;                                      // tensor_dim0_stride
        g1[6] = 0; g1[7] = 0;
        i32x4 g2 = (i32x4)0, g3 = (i32x4)0;
#if __clang_major__ >= 23
        i32x8 g4 = (i32x8)0;
        __builtin_amdgcn_tensor_load_to_lds(g0, g1, g2, g3, g4, 0);
#else
        __builtin_amdgcn_tensor_load_to_lds(g0, g1, g2, g3, 0);
#endif
        __builtin_amdgcn_s_wait_tensorcnt(0);
      }
#else
      // bf16 N x K: 16 contiguous k (32B) per thread -> aligned b128 in and out
      const u32x4* src = (const u32x4*)(Bb + (size_t)(colBase + tn1) * (size_t)ldb + k0 + tk1);
      u32x4 c0 = src[0], c1 = src[1];
      u32x4* dst = (u32x4*)&Bs[tn1 * LP + tk1];
      dst[0] = c0; dst[1] = c1;
#endif
    } else {
      // bf16 K x N with N guard (P@V, N=64)
      #pragma unroll
      for (int e = 0; e < 16; e++) {
        int n = bn + e, ng = colBase + n;
        float v = 0.0f;
        if (!NG || ng < N) v = (float)Bb[(size_t)(k0 + bk) * (size_t)ldb + ng];
        Bs[n * LP + bk] = (__bf16)v;
      }
    }
    __syncthreads();

    // ---- fragments per documented 16-bit WMMA VGPR layouts ----
    union Frag { u32x4 u[2]; v16bf v; };
    Frag a[2], b[4];
    #pragma unroll
    for (int tm = 0; tm < 2; tm++) {
      int row = wm * 32 + tm * 16 + l16;
      const u32x4* p = (const u32x4*)&As[row * LP + 8 * half];
      a[tm].u[0] = p[0];
      a[tm].u[1] = p[2];           // +32B = +16 halves
    }
    #pragma unroll
    for (int tn = 0; tn < 4; tn++) {
      int col = wn * 64 + tn * 16 + l16;
      const u32x4* p = (const u32x4*)&Bs[col * LP + 8 * half];
      b[tn].u[0] = p[0];
      b[tn].u[1] = p[2];
    }
    #pragma unroll
    for (int tm = 0; tm < 2; tm++)
      #pragma unroll
      for (int tn = 0; tn < 4; tn++)
        acc[tm][tn] = __builtin_amdgcn_wmma_f32_16x16x32_bf16(
            false, a[tm].v, false, b[tn].v, (short)0, acc[tm][tn], false, false);
    __syncthreads();
  }

  // ---- epilogue: bias/scale/relu, fp32 and/or bf16 stores ----
  #pragma unroll
  for (int tm = 0; tm < 2; tm++) {
    #pragma unroll
    for (int tn = 0; tn < 4; tn++) {
      int ng = colBase + wn * 64 + tn * 16 + l16;
      if (NG && ng >= N) continue;
      float bi = bias ? bias[ng] : 0.0f;
      #pragma unroll
      for (int r = 0; r < 8; r++) {
        int mg = rowBase + wm * 32 + tm * 16 + r + 8 * half;  // C layout: VGPR r -> M=r(+8)
        float v = acc[tm][tn][r] * scale + bi;
        if (relu) v = v > 0.0f ? v : 0.0f;
        if (Cf) Cf[(size_t)mg * ldcf + ng] = v;
        if (Cb) {
          size_t o;
          if constexpr (QKV) {   // [B,H,S,DEPTH] head-major scatter
            int b_ = mg >> 11, s_ = mg & (SS - 1);
            int h_ = ng >> 6,  d_ = ng & (DEPTH - 1);
            o = ((size_t)(b_ * HH + h_) << 17) + ((size_t)s_ << 6) + d_;
          } else {
            o = (size_t)mg * ldcb + ng;
          }
          Cb[o] = (__bf16)v;
        }
      }
    }
  }
}

// ============================================================================
// Row softmax: one block per row of `ncols` fp32, bf16 output
// ============================================================================
__global__ __launch_bounds__(256) void softmax_rows(const float* __restrict__ S,
                                                    __bf16* __restrict__ P, int ncols)
{
  __shared__ float red[256];
  const int row = blockIdx.x, tid = threadIdx.x;
  const float* x = S + (size_t)row * ncols;
  const int per = ncols / 256;   // 8
  float r[8];
  float m = -1e30f;
  for (int i = 0; i < per; i++) { r[i] = x[tid + i * 256]; m = fmaxf(m, r[i]); }
  red[tid] = m; __syncthreads();
  for (int s = 128; s > 0; s >>= 1) { if (tid < s) red[tid] = fmaxf(red[tid], red[tid + s]); __syncthreads(); }
  m = red[0]; __syncthreads();
  float sum = 0.0f;
  for (int i = 0; i < per; i++) { r[i] = __expf(r[i] - m); sum += r[i]; }
  red[tid] = sum; __syncthreads();
  for (int s = 128; s > 0; s >>= 1) { if (tid < s) red[tid] += red[tid + s]; __syncthreads(); }
  float inv = 1.0f / red[0];
  __bf16* y = P + (size_t)row * ncols;
  for (int i = 0; i < per; i++) y[tid + i * 256] = (__bf16)(r[i] * inv);
}

// ============================================================================
// Fused residual-add + LayerNorm: out = LN(X + Yd)*g + be ; fp32 + optional bf16
// ============================================================================
__global__ __launch_bounds__(256) void ln_residual(const float* __restrict__ X,
                                                   const float* __restrict__ Yd,
                                                   const float* __restrict__ g,
                                                   const float* __restrict__ be,
                                                   float* __restrict__ outF,
                                                   __bf16* __restrict__ outB, int ncols)
{
  __shared__ float red[256];
  const int row = blockIdx.x, tid = threadIdx.x;
  const float* xp = X + (size_t)row * ncols;
  const float* yp = Yd + (size_t)row * ncols;
  const int per = ncols / 256;   // 4
  float v[4];
  float s = 0.0f;
  for (int i = 0; i < per; i++) { v[i] = xp[tid + i * 256] + yp[tid + i * 256]; s += v[i]; }
  red[tid] = s; __syncthreads();
  for (int t = 128; t > 0; t >>= 1) { if (tid < t) red[tid] += red[tid + t]; __syncthreads(); }
  float mean = red[0] / (float)ncols; __syncthreads();
  s = 0.0f;
  for (int i = 0; i < per; i++) { float d = v[i] - mean; s += d * d; }
  red[tid] = s; __syncthreads();
  for (int t = 128; t > 0; t >>= 1) { if (tid < t) red[tid] += red[tid + t]; __syncthreads(); }
  float rstd = rsqrtf(red[0] / (float)ncols + 1e-6f);
  for (int i = 0; i < per; i++) {
    int c = tid + i * 256;
    float o = (v[i] - mean) * rstd * g[c] + be[c];
    if (outF) outF[(size_t)row * ncols + c] = o;
    if (outB) outB[(size_t)row * ncols + c] = (__bf16)o;
  }
}

// ============================================================================
// Embedding * sqrt(D) + positional encoding (concat(sin(even), cos(odd)))
// ============================================================================
__global__ __launch_bounds__(256) void embed_pos(const int* __restrict__ x,
                                                 const float* __restrict__ emb,
                                                 float* __restrict__ Hf,
                                                 __bf16* __restrict__ Hb)
{
  int idx = blockIdx.x * 256 + threadIdx.x;   // over B*S*D exactly
  int d  = idx & (DD - 1);
  int bs = idx >> 10;
  int s  = bs & (SS - 1);
  int tok = x[bs];
  float pe;
  if (d < DD / 2) {
    float ang = (float)s * __powf(10000.0f, -(2.0f * (float)d) / (float)DD);
    pe = __sinf(ang);
  } else {
    float ang = (float)s * __powf(10000.0f, -(2.0f * (float)(d - DD / 2)) / (float)DD);
    pe = __cosf(ang);
  }
  float v = emb[(size_t)tok * DD + d] * 32.0f + pe;   // sqrt(1024) = 32
  Hf[idx] = v;
  Hb[idx] = (__bf16)v;
}

// ============================================================================
// Host orchestration
// ============================================================================
extern "C" void kernel_launch(void* const* d_in, const int* in_sizes, int n_in,
                              void* d_out, int out_size, void* d_ws, size_t ws_size,
                              hipStream_t stream) {
  const int*   x    = (const int*)  d_in[0];
  const float* emb  = (const float*)d_in[1];
  const float* Wq   = (const float*)d_in[2];
  const float* bq   = (const float*)d_in[3];
  const float* Wk   = (const float*)d_in[4];
  const float* bk   = (const float*)d_in[5];
  const float* Wv   = (const float*)d_in[6];
  const float* bv   = (const float*)d_in[7];
  const float* Wo   = (const float*)d_in[8];
  const float* bo   = (const float*)d_in[9];
  const float* g1   = (const float*)d_in[10];
  const float* be1  = (const float*)d_in[11];
  const float* W1   = (const float*)d_in[12];
  const float* c1   = (const float*)d_in[13];
  const float* W2   = (const float*)d_in[14];
  const float* c2   = (const float*)d_in[15];
  const float* g2   = (const float*)d_in[16];
  const float* be2  = (const float*)d_in[17];

  // ---- workspace carve-out ----
  size_t off = 0;
  auto alloc = [&](size_t bytes) -> char* {
    char* p = (char*)d_ws + off;
    off = (off + bytes + 255) & ~(size_t)255;
    return p;
  };
  float*  Hf  = (float*) alloc((size_t)MM * DD * 4);            // activations fp32
  __bf16* Hb  = (__bf16*)alloc((size_t)MM * DD * 2);            // activations bf16
  __bf16* Qb  = (__bf16*)alloc((size_t)BB * HH * SS * DEPTH * 2);
  __bf16* Kb  = (__bf16*)alloc((size_t)BB * HH * SS * DEPTH * 2);
  __bf16* Vb  = (__bf16*)alloc((size_t)BB * HH * SS * DEPTH * 2);
  float*  Sc  = (float*) alloc((size_t)SS * SS * 4);            // scores (one head)
  __bf16* Pb  = (__bf16*)alloc((size_t)SS * SS * 2);            // probs  (one head)
  __bf16* Ob  = (__bf16*)alloc((size_t)MM * DD * 2);            // attn concat bf16
  float*  Of  = (float*) alloc((size_t)MM * DD * 4);            // O-projection fp32
  float*  H1f = (float*) alloc((size_t)MM * DD * 4);
  __bf16* H1b = (__bf16*)alloc((size_t)MM * DD * 2);
  __bf16* Mb  = (__bf16*)alloc((size_t)MM * DFF * 2);           // FFN mid bf16
  float*  Ff  = (float*) alloc((size_t)MM * DD * 4);

  // ---- embedding + positional encoding ----
  embed_pos<<<(MM * DD) / 256, 256, 0, stream>>>(x, emb, Hf, Hb);

  const float inv_sqrt_depth = 0.125f;   // 1/sqrt(64)
  const dim3 blk(256);

  for (int l = 0; l < LL; l++) {
    const float* Wq_l = Wq + (size_t)l * DD * DD;
    const float* Wk_l = Wk + (size_t)l * DD * DD;
    const float* Wv_l = Wv + (size_t)l * DD * DD;
    const float* Wo_l = Wo + (size_t)l * DD * DD;
    const float* W1_l = W1 + (size_t)l * DD * DFF;
    const float* W2_l = W2 + (size_t)l * DFF * DD;

    // Q/K/V projections -> bf16 head-major [B,H,S,DEPTH]
    {
      dim3 grid(DD / 128, MM / 128);
      gemm_wmma_bf16<0, false, true><<<grid, blk, 0, stream>>>(
          Hb, DD, Wq_l, DD, bq + (size_t)l * DD, nullptr, 0, Qb, 0, MM, DD, DD, 1.0f, 0);
      gemm_wmma_bf16<0, false, true><<<grid, blk, 0, stream>>>(
          Hb, DD, Wk_l, DD, bk + (size_t)l * DD, nullptr, 0, Kb, 0, MM, DD, DD, 1.0f, 0);
      gemm_wmma_bf16<0, false, true><<<grid, blk, 0, stream>>>(
          Hb, DD, Wv_l, DD, bv + (size_t)l * DD, nullptr, 0, Vb, 0, MM, DD, DD, 1.0f, 0);
    }

    // attention, one (b,h) pair at a time (bounds workspace to 24 MB)
    for (int bh = 0; bh < BB * HH; bh++) {
      int b_ = bh / HH, h_ = bh % HH;
      const __bf16* Qh = Qb + (size_t)bh * SS * DEPTH;
      const __bf16* Kh = Kb + (size_t)bh * SS * DEPTH;
      const __bf16* Vh = Vb + (size_t)bh * SS * DEPTH;
      // scores = Q @ K^T / 8 : B is bf16 N x K row-major (TDM tile load path)
      {
        dim3 grid(SS / 128, SS / 128);
        gemm_wmma_bf16<1, false, false><<<grid, blk, 0, stream>>>(
            Qh, DEPTH, Kh, DEPTH, nullptr, Sc, SS, nullptr, 0,
            SS, SS, DEPTH, inv_sqrt_depth, 0);
      }
      softmax_rows<<<SS, 256, 0, stream>>>(Sc, Pb, SS);
      // O_h = P @ V : bf16 K x N row-major, N=64 (guarded)
      {
        dim3 grid(1, SS / 128);
        gemm_wmma_bf16<2, true, false><<<grid, blk, 0, stream>>>(
            Pb, SS, Vh, DEPTH, nullptr, nullptr, 0,
            Ob + (size_t)b_ * SS * DD + (size_t)h_ * DEPTH, DD,
            SS, DEPTH, SS, 1.0f, 0);
      }
    }

    // output projection + residual LN
    {
      dim3 grid(DD / 128, MM / 128);
      gemm_wmma_bf16<0, false, false><<<grid, blk, 0, stream>>>(
          Ob, DD, Wo_l, DD, bo + (size_t)l * DD, Of, DD, nullptr, 0, MM, DD, DD, 1.0f, 0);
    }
    ln_residual<<<MM, 256, 0, stream>>>(Hf, Of, g1 + (size_t)l * DD, be1 + (size_t)l * DD,
                                        H1f, H1b, DD);

    // FFN
    {
      dim3 grid(DFF / 128, MM / 128);
      gemm_wmma_bf16<0, false, false><<<grid, blk, 0, stream>>>(
          H1b, DD, W1_l, DFF, c1 + (size_t)l * DFF, nullptr, 0, Mb, DFF, MM, DFF, DD, 1.0f, 1);
    }
    {
      dim3 grid(DD / 128, MM / 128);
      gemm_wmma_bf16<0, false, false><<<grid, blk, 0, stream>>>(
          Mb, DFF, W2_l, DD, c2 + (size_t)l * DD, Ff, DD, nullptr, 0, MM, DD, DFF, 1.0f, 0);
    }

    float* h2out = (l == LL - 1) ? (float*)d_out : Hf;
    ln_residual<<<MM, 256, 0, stream>>>(H1f, Ff, g2 + (size_t)l * DD, be2 + (size_t)l * DD,
                                        h2out, Hb, DD);
  }
}
